// EnhancedTemporalGNN_76836964926217
// MI455X (gfx1250) — matrix operand order repaired and checked by reading, not compiled, not run on hardware
//
#include <hip/hip_runtime.h>
#include <hip/hip_bf16.h>
#include <math.h>

#define DEV __device__ __forceinline__

typedef __attribute__((ext_vector_type(16))) __bf16 v16bf;
typedef __attribute__((ext_vector_type(8)))  float  v8f;

namespace {

constexpr int kN  = 20000;
constexpr int kE  = 80000;
constexpr int kD  = 128;
constexpr int kH  = 8;     // layer attention heads (hd=16)
constexpr int kW  = 6;     // trust evolution window
constexpr int kL  = 2;
constexpr int kED = 64;    // edge msg dim
constexpr int kCH = 8192;  // trust-pipeline edge chunk

// ---------------- helpers ----------------

DEV __bf16 f2bf(float f) {
  union { float f; unsigned u; } in; in.f = f;
  unsigned r = in.u + 0x7FFFu + ((in.u >> 16) & 1u);   // round-to-nearest-even
  union { unsigned short s; __bf16 b; } out; out.s = (unsigned short)(r >> 16);
  return out.b;
}

DEV float gelu_exact(float x) { return 0.5f * x * (1.0f + erff(x * 0.70710678118f)); }

// monotonic unsigned encoding of float for atomicMax over signed values
DEV unsigned fenc(float f) {
  unsigned u = __float_as_uint(f);
  return (u & 0x80000000u) ? ~u : (u | 0x80000000u);
}
DEV float fdec(unsigned u) {
  return (u & 0x80000000u) ? __uint_as_float(u & 0x7FFFFFFFu) : __uint_as_float(~u);
}

union FragU { uint4 q[2]; v16bf v; };

// ---------------- weight pack: fragment-order bf16 ----------------
// Wp[((ct*(K/32)+kc)*32 + lane)*16 + i] = bf16( Wm[(kc*32 + 8*(lane>>4) + (i&7) + 16*(i>>3))*ncols + ct*16 + (lane&15)] )
__global__ void k_packw(const float* __restrict__ Wm, int K, int ncols,
                        __bf16* __restrict__ Wp) {
  const int KC32 = K >> 5;
  long i = (long)blockIdx.x * blockDim.x + threadIdx.x;
  const long tot = (long)K * ncols;
  for (; i < tot; i += (long)gridDim.x * blockDim.x) {
    int e    = (int)(i & 15);
    int l    = (int)((i >> 4) & 31);
    int rest = (int)(i >> 9);
    int kc   = rest % KC32;
    int ct   = rest / KC32;
    int row  = kc * 32 + ((l >> 4) << 3) + (e & 7) + ((e >> 3) << 4);
    int col  = ct * 16 + (l & 15);
    Wp[i] = f2bf(Wm[(long)row * ncols + col]);
  }
}

// ---------------- generic gathered bf16 WMMA GEMM ----------------
// C/Cb[r,:] = act( gather(A)[r,:KC] @ W[KC,ncols] + bias )  (accum: += prior fp32 C)
// gmode 0: row r; 1: row gidx[r]; 2: evo-window row gidx[r/6]*6 + r%6
// Block = 128 threads = 4 waves; each wave owns ONE 16-row M-tile (A loaded once
// per row, no duplication across waves) and sweeps ALL NTILES col-tiles with a
// single live accumulator tile. B comes from the packed (L2-resident) buffer.
template <int KC, int NTILES>
__global__ __launch_bounds__(128) void k_gemm(
    const __bf16* __restrict__ A, const int* __restrict__ gidx, int gmode,
    int M, int ldA,
    const __bf16* __restrict__ Wp, int ncols,
    const float* __restrict__ bias,
    float* __restrict__ C, __bf16* __restrict__ Cb,
    int act, int accum)
{
  constexpr int KC32 = KC / 32;
  const int lane  = threadIdx.x & 31;
  const int wid   = threadIdx.x >> 5;
  const int mBase = ((int)blockIdx.x * 4 + wid) * 16;
  if (mBase >= M) return;   // wave-uniform exit

  int rA = mBase + (lane & 15);
  if (rA >= M) rA = M - 1;
  long srcRow;
  if (gmode == 0)      srcRow = rA;
  else if (gmode == 1) srcRow = gidx[rA];
  else                 srcRow = (long)gidx[rA / kW] * kW + (rA % kW);
  const __bf16* Arow = A + srcRow * (long)ldA;

  // A fragments: elems 0-7 -> K kb..kb+7, elems 8-15 -> K kb+16..kb+23 (ISA 16-bit A layout)
  v16bf af[KC32];
  #pragma unroll
  for (int kc = 0; kc < KC32; ++kc) {
    const int kb = kc * 32 + ((lane >> 4) << 3);
    FragU u;
    u.q[0] = *(const uint4*)(Arow + kb);
    u.q[1] = *(const uint4*)(Arow + kb + 16);
    af[kc] = u.v;
  }

  const int colLane = lane & 15;
  const int rowOff  = (lane >> 4) << 3;

  #pragma unroll
  for (int ct = 0; ct < NTILES; ++ct) {
    const int col = ct * 16 + colLane;
    v8f acc = {0.f, 0.f, 0.f, 0.f, 0.f, 0.f, 0.f, 0.f};
    if (accum) {
      #pragma unroll
      for (int v = 0; v < 8; ++v) {
        int row = mBase + v + rowOff;
        acc[v] = (row < M) ? C[(long)row * ncols + col] : 0.f;
      }
    }
    #pragma unroll
    for (int kc = 0; kc < KC32; ++kc) {
      const __bf16* bp = Wp + (((long)ct * KC32 + kc) * 32 + lane) * 16;
      FragU ub;
      ub.q[0] = *(const uint4*)bp;
      ub.q[1] = *(const uint4*)(bp + 8);
      acc = __builtin_amdgcn_wmma_f32_16x16x32_bf16(
          false, af[kc], false, ub.v, (short)0, acc, false, false);
    }
    const float b = bias ? bias[col] : 0.f;
    #pragma unroll
    for (int v = 0; v < 8; ++v) {
      int row = mBase + v + rowOff;
      if (row < M) {
        float val = acc[v] + b;
        if (act == 1) val = gelu_exact(val);
        if (C)  C[(long)row * ncols + col]  = val;
        if (Cb) Cb[(long)row * ncols + col] = f2bf(val);
      }
    }
  }
}

// ---------------- small elementwise / edge kernels ----------------

__global__ void k_fill(float* p, float v, long n) {
  long i = (long)blockIdx.x * blockDim.x + threadIdx.x;
  for (; i < n; i += (long)gridDim.x * blockDim.x) p[i] = v;
}

__global__ void k_cvt(const float* __restrict__ in, __bf16* __restrict__ out, long n) {
  long i = (long)blockIdx.x * blockDim.x + threadIdx.x;
  for (; i < n; i += (long)gridDim.x * blockDim.x) out[i] = f2bf(in[i]);
}

__global__ void k_segmax(const float* __restrict__ t, const int* __restrict__ src,
                         unsigned* __restrict__ last) {
  long i = (long)blockIdx.x * blockDim.x + threadIdx.x;
  for (; i < kE; i += (long)gridDim.x * blockDim.x)
    atomicMax(&last[src[i]], __float_as_uint(t[i]));   // t >= 0, init 0 == maximum(.,0)
}

__global__ void k_trel(const float* __restrict__ t, const int* __restrict__ src,
                       const float* __restrict__ last, float* __restrict__ trel) {
  long i = (long)blockIdx.x * blockDim.x + threadIdx.x;
  for (; i < kE; i += (long)gridDim.x * blockDim.x) trel[i] = t[i] - last[src[i]];
}

// time2vec features straight to bf16 (only a GEMM consumes them)
__global__ void k_time_feats(const float* __restrict__ t, const float* __restrict__ trel,
    const float* w0, const float* b0, const float* Wf, const float* Bf,
    const float* rw0, const float* rb0, const float* rWf, const float* rBf,
    __bf16* __restrict__ c)
{
  long i = (long)blockIdx.x * blockDim.x + threadIdx.x;
  const long tot = (long)kE * kD;
  for (; i < tot; i += (long)gridDim.x * blockDim.x) {
    int e = (int)(i >> 7), j = (int)(i & 127);
    float v;
    if (j == 0)       v = t[e] * w0[0] + b0[0];
    else if (j < 64)  v = sinf(t[e] * Wf[j - 1] + Bf[j - 1]);
    else if (j == 64) v = trel[e] * rw0[0] + rb0[0];
    else              v = sinf(trel[e] * rWf[j - 65] + rBf[j - 65]);
    c[i] = f2bf(v);
  }
}

// trust attention over window W=6 (4 heads, hd=32); one wave per edge in chunk
__global__ __launch_bounds__(256) void k_trust_attn(
    const float* __restrict__ Qt, const float* __restrict__ Kt,
    const float* __restrict__ Vt, const float* __restrict__ log_decay,
    __bf16* __restrict__ pooled, int chunkStart, int mc)
{
  const int lane = threadIdx.x & 31;
  const int el   = blockIdx.x * 8 + (threadIdx.x >> 5);
  if (el >= mc) return;
  const int h  = lane >> 3;                 // 4 trust heads, 8 lanes each
  const int d0 = h * 32 + (lane & 7) * 4;   // 4 dims per lane
  const int eg = chunkStart + el;
  const float scale = 0.17677669529f;       // 32^-0.5

  float q[4];
  #pragma unroll
  for (int k = 0; k < 4; ++k) q[k] = Qt[(long)eg * kD + d0 + k];

  float dec = expf(log_decay[0]);
  float dw[kW]; float dsum = 0.f;
  #pragma unroll
  for (int w = 0; w < kW; ++w) { dw[w] = expf(dec * (float)w); dsum += dw[w]; }

  float sc[kW];
  #pragma unroll
  for (int w = 0; w < kW; ++w) {
    const float* kr = Kt + ((long)el * kW + w) * kD + d0;
    float p = q[0]*kr[0] + q[1]*kr[1] + q[2]*kr[2] + q[3]*kr[3];
    p += __shfl_xor(p, 1); p += __shfl_xor(p, 2); p += __shfl_xor(p, 4);
    sc[w] = p * scale + dw[w] / dsum;
  }
  float mx = sc[0];
  #pragma unroll
  for (int w = 1; w < kW; ++w) mx = fmaxf(mx, sc[w]);
  float ssum = 0.f;
  #pragma unroll
  for (int w = 0; w < kW; ++w) { sc[w] = expf(sc[w] - mx); ssum += sc[w]; }
  const float inv = 1.f / ssum;
  float out[4] = {0.f, 0.f, 0.f, 0.f};
  #pragma unroll
  for (int w = 0; w < kW; ++w) {
    const float* vr = Vt + ((long)el * kW + w) * kD + d0;
    float a = sc[w] * inv;
    out[0] += a*vr[0]; out[1] += a*vr[1]; out[2] += a*vr[2]; out[3] += a*vr[3];
  }
  #pragma unroll
  for (int k = 0; k < 4; ++k) pooled[(long)eg * kD + d0 + k] = f2bf(out[k]);
}

// trust logit + confidence -> sigmoid; one wave per edge
__global__ __launch_bounds__(256) void k_trust_fin(
    const float* __restrict__ to1o, const float* __restrict__ to2w, const float* to2b,
    const float* __restrict__ mem_v, const int* __restrict__ src, const int* __restrict__ dst,
    float* __restrict__ trust)
{
  const int lane = threadIdx.x & 31;
  const int e    = blockIdx.x * 8 + (threadIdx.x >> 5);
  if (e >= kE) return;
  float lg = to1o[(long)e*64 + lane] * to2w[lane]
           + to1o[(long)e*64 + 32 + lane] * to2w[32 + lane];
  float vi = 0.f, vj = 0.f;
  const float* ri = mem_v + (long)dst[e] * kD;
  const float* rj = mem_v + (long)src[e] * kD;
  #pragma unroll
  for (int k = 0; k < 4; ++k) { vi += ri[lane + 32*k]; vj += rj[lane + 32*k]; }
  for (int m = 1; m < 32; m <<= 1) {
    lg += __shfl_xor(lg, m); vi += __shfl_xor(vi, m); vj += __shfl_xor(vj, m);
  }
  if (lane == 0) {
    lg += to2b[0];
    float ci = 1.f / (1.f + fmaxf(sqrtf(vi / 128.f), 1e-6f));
    float cj = 1.f / (1.f + fmaxf(sqrtf(vj / 128.f), 1e-6f));
    float conf = sqrtf(ci * cj);
    float z = lg + fmaxf(logf(conf), -5.f);
    trust[e] = 1.f / (1.f + expf(-z));
  }
}

// per-edge attention logits (H=8, hd=16) * trust, plus segment max; one wave per edge
__global__ __launch_bounds__(256) void k_edge_attn(
    const float* __restrict__ Qn, const float* __restrict__ Kn,
    const float* __restrict__ eeP, const float* __restrict__ teP,
    const float* __restrict__ trust, const int* __restrict__ src,
    const int* __restrict__ dst, float* __restrict__ attn, unsigned* __restrict__ menc)
{
  const int lane = threadIdx.x & 31;
  const int e    = blockIdx.x * 8 + (threadIdx.x >> 5);
  if (e >= kE) return;
  const int h  = lane >> 2;               // 8 heads, 4 lanes each
  const int d0 = h * 16 + (lane & 3) * 4;
  const float* qr = Qn  + (long)dst[e] * kD + d0;
  const float* kr = Kn  + (long)src[e] * kD + d0;
  const float* er = eeP + (long)e * kD + d0;
  const float* tr = teP + (long)e * kD + d0;
  float p = 0.f;
  #pragma unroll
  for (int k = 0; k < 4; ++k) p += qr[k] * (kr[k] + er[k] + tr[k]);
  p += __shfl_xor(p, 1); p += __shfl_xor(p, 2);
  if ((lane & 3) == 0) {
    float a = p * 0.25f * trust[e];       // scale = 16^-0.5
    attn[(long)e * kH + h] = a;
    atomicMax(&menc[(long)dst[e] * kH + h], fenc(a));
  }
}

__global__ void k_edge_exp(float* __restrict__ attn, const unsigned* __restrict__ menc,
                           float* __restrict__ s, const int* __restrict__ dst) {
  long i = (long)blockIdx.x * blockDim.x + threadIdx.x;
  const long tot = (long)kE * kH;
  for (; i < tot; i += (long)gridDim.x * blockDim.x) {
    int e = (int)(i >> 3), h = (int)(i & 7);
    float m  = fdec(menc[(long)dst[e] * kH + h]);
    float ex = expf(attn[i] - m);
    attn[i] = ex;
    atomicAdd(&s[(long)dst[e] * kH + h], ex);
  }
}

__global__ __launch_bounds__(256) void k_edge_agg(
    const float* __restrict__ ex, const float* __restrict__ s,
    const float* __restrict__ Vn, const int* __restrict__ src,
    const int* __restrict__ dst, float* __restrict__ agg)
{
  const int lane = threadIdx.x & 31;
  const int e    = blockIdx.x * 8 + (threadIdx.x >> 5);
  if (e >= kE) return;
  const int sd = dst[e], ss = src[e];
  #pragma unroll
  for (int k = 0; k < 4; ++k) {
    int d = lane + 32 * k;
    int h = d >> 4;
    float a = ex[(long)e * kH + h] / (s[(long)sd * kH + h] + 1e-16f);
    atomicAdd(&agg[(long)sd * kD + d], Vn[(long)ss * kD + d] * a);
  }
}

// gate + residual + layernorm (gw==null -> plain layernorm of x); one wave per node
__global__ __launch_bounds__(256) void k_gate_ln(
    const float* __restrict__ x, const float* __restrict__ o,
    const float* gw, const float* gb,
    const float* __restrict__ gamma, const float* __restrict__ beta,
    float* __restrict__ out, __bf16* __restrict__ outb, int n)
{
  const int lane = threadIdx.x & 31;
  const int i    = blockIdx.x * 8 + (threadIdx.x >> 5);
  if (i >= n) return;
  const float* xr = x + (long)i * kD;
  float y[4];
  if (gw) {
    float gs = 0.f;
    #pragma unroll
    for (int k = 0; k < 4; ++k) gs += xr[lane + 32*k] * gw[lane + 32*k];
    for (int m = 1; m < 32; m <<= 1) gs += __shfl_xor(gs, m);
    float g = 1.f / (1.f + expf(-(gs + gb[0])));
    const float* orow = o + (long)i * kD;
    #pragma unroll
    for (int k = 0; k < 4; ++k) { int d = lane + 32*k; y[k] = g * xr[d] + (1.f - g) * orow[d]; }
  } else {
    #pragma unroll
    for (int k = 0; k < 4; ++k) y[k] = xr[lane + 32*k];
  }
  float sm = y[0] + y[1] + y[2] + y[3];
  for (int m = 1; m < 32; m <<= 1) sm += __shfl_xor(sm, m);
  const float mean = sm / 128.f;
  float vs = 0.f;
  #pragma unroll
  for (int k = 0; k < 4; ++k) { float d = y[k] - mean; vs += d * d; }
  for (int m = 1; m < 32; m <<= 1) vs += __shfl_xor(vs, m);
  const float inv = rsqrtf(vs / 128.f + 1e-5f);
  #pragma unroll
  for (int k = 0; k < 4; ++k) {
    int d = lane + 32 * k;
    float r = (y[k] - mean) * inv * gamma[d] + beta[d];
    out[(long)i * kD + d] = r;
    if (outb) outb[(long)i * kD + d] = f2bf(r);
  }
}

struct LayerP {
  const float *e_b,*e_w,*f1_b,*f1_w,*f2_b,*f2_w,*g_b,*g_w,*k_b,*k_w,*log_decay,
              *n_b,*n_g,*o_b,*o_w,*q_b,*q_w,*tk_b,*tk_w,*tm_b,*tm_w,
              *to1_b,*to1_w,*to2_b,*to2_w,*tq_b,*tq_w,*tv_b,*tv_w,*v_b,*v_w;
};

} // namespace

extern "C" void kernel_launch(void* const* d_in, const int* in_sizes, int n_in,
                              void* d_out, int out_size, void* d_ws, size_t ws_size,
                              hipStream_t stream) {
  (void)in_sizes; (void)out_size; (void)ws_size;
  const float* x   = (const float*)d_in[0];
  const int*   ei  = (const int*)  d_in[1];
  const float* msg = (const float*)d_in[2];
  const float* tt  = (const float*)d_in[3];
  const int* src = ei;
  const int* dst = ei + kE;

  // params flattened jax-style (dicts sorted by key, lists in order), after inputs:
  //  0 ee_b 1 ee_w 2 evo 3 fn_b 4 fn_g 5 in_b 6 in_w, 7..37 layer0, 38..68 layer1,
  //  69 mem_h 70 mem_v, 71..82 time (B,W,b0,p1b,p1w,p2b,p2w,rB,rW,rb0,rw0,w0)
  const float* Pp[83];
  for (int i = 0; i < 83; ++i) {
    int idx = 4 + i; if (idx >= n_in) idx = n_in - 1;
    Pp[i] = (const float*)d_in[idx];
  }
  const float *ee_b = Pp[0], *ee_w = Pp[1], *evo = Pp[2], *fn_b = Pp[3], *fn_g = Pp[4];
  const float *in_b = Pp[5], *in_w = Pp[6], *mem_h = Pp[69], *mem_v = Pp[70];
  const float *tB = Pp[71], *tW = Pp[72], *tb0 = Pp[73], *p1b = Pp[74], *p1w = Pp[75];
  const float *p2b = Pp[76], *p2w = Pp[77], *rB = Pp[78], *rW = Pp[79];
  const float *rb0 = Pp[80], *rw0 = Pp[81], *tw0 = Pp[82];
  LayerP LP[kL];
  for (int l = 0; l < kL; ++l) {
    const float** f = (const float**)&LP[l];
    for (int i = 0; i < 31; ++i) f[i] = Pp[7 + 31 * l + i];
  }

  // ----- workspace (byte allocator, 64B aligned blocks) -----
  char* wsb = (char*)d_ws; size_t off = 0;
  auto allocB = [&](size_t bytes) { void* p = wsb + off; off = (off + bytes + 63) & ~(size_t)63; return p; };
  auto allocF = [&](size_t n) { return (float*)allocB(n * 4); };
  auto allocH = [&](size_t n) { return (__bf16*)allocB(n * 2); };

  // fp32 buffers
  float* last = allocF(kN);
  float* trel = allocF(kE);
  float* hA   = allocF((size_t)kN * kD);
  float* hB   = allocF((size_t)kN * kD);
  float* Qn   = allocF((size_t)kN * kD);
  float* Kn   = allocF((size_t)kN * kD);
  float* Vn   = allocF((size_t)kN * kD);
  float* eeP  = allocF((size_t)kE * kD);
  float* teP  = allocF((size_t)kE * kD);
  float* Qt   = allocF((size_t)kE * kD);
  float* Ktc  = allocF((size_t)kCH * kW * kD);
  float* Vtc  = allocF((size_t)kCH * kW * kD);
  float* trust= allocF(kE);
  float* attn = allocF((size_t)kE * kH);
  float* mB   = allocF((size_t)kN * kH);
  float* sB   = allocF((size_t)kN * kH);
  float* agg  = allocF((size_t)kN * kD);
  float* fu1  = allocF((size_t)kN * kD);
  float* fu2  = allocF((size_t)kN * kD);
  float* to1o = Ktc;   // alias: Ktc dead when to1 runs (E*64 <= kCH*6*128)
  // bf16 buffers
  __bf16* xb    = allocH((size_t)kN * kD);
  __bf16* msgb  = allocH((size_t)kE * kED);
  __bf16* evob  = allocH((size_t)kN * kW * kD);
  __bf16* memhb = allocH((size_t)kN * kD);
  __bf16* cb    = allocH((size_t)kE * kD);    // time features
  __bf16* tmpb  = allocH((size_t)kE * kD);    // p1 out; reused as trust-pooled
  __bf16* timEb = allocH((size_t)kE * kD);
  __bf16* edgEb = allocH((size_t)kE * kD);
  __bf16* hb    = allocH((size_t)kN * kD);
  __bf16* aggb  = allocH((size_t)kN * kD);
  __bf16* fu1b  = allocH((size_t)kN * kD);
  __bf16* fu2b  = allocH((size_t)kN * kD);
  __bf16* wpack = allocH((size_t)256 * 128);  // packed weight fragments (max 256x128)
  __bf16* poolb = tmpb;

  const int GS = 1024, BS = 256;
  auto waves = [](int n) { return (n + 7) / 8; };

  auto gemm = [&](const __bf16* A, const int* gidx, int gmode, int M, int K, int ldA,
                  const float* Wm, int ncols, const float* bias,
                  float* C, __bf16* Cb, int act, int accum) {
    k_packw<<<64, BS, 0, stream>>>(Wm, K, ncols, wpack);
    dim3 g((M + 63) / 64);     // 4 waves/block, one 16-row M-tile per wave
    if (K == 64)
      k_gemm<64, 8><<<g, 128, 0, stream>>>(A, gidx, gmode, M, ldA, wpack, ncols, bias, C, Cb, act, accum);
    else if (ncols == 128)
      k_gemm<128, 8><<<g, 128, 0, stream>>>(A, gidx, gmode, M, ldA, wpack, ncols, bias, C, Cb, act, accum);
    else
      k_gemm<128, 4><<<g, 128, 0, stream>>>(A, gidx, gmode, M, ldA, wpack, ncols, bias, C, Cb, act, accum);
  };

  // ----- one-time bf16 conversions of GEMM A-inputs -----
  k_cvt<<<GS, BS, 0, stream>>>(x,     xb,    (long)kN * kD);
  k_cvt<<<GS, BS, 0, stream>>>(msg,   msgb,  (long)kE * kED);
  k_cvt<<<2048, BS, 0, stream>>>(evo, evob,  (long)kN * kW * kD);
  k_cvt<<<GS, BS, 0, stream>>>(mem_h, memhb, (long)kN * kD);

  // ----- front end -----
  k_fill<<<GS, BS, 0, stream>>>(last, 0.f, kN);
  k_segmax<<<GS, BS, 0, stream>>>(tt, src, (unsigned*)last);
  k_trel<<<GS, BS, 0, stream>>>(tt, src, last, trel);
  k_time_feats<<<GS, BS, 0, stream>>>(tt, trel, tw0, tb0, tW, tB, rw0, rb0, rW, rB, cb);
  gemm(cb,   nullptr, 0, kE, kD, kD, p1w, kD, p1b, nullptr, tmpb, /*gelu*/1, 0);
  gemm(tmpb, nullptr, 0, kE, kD, kD, p2w, kD, p2b, nullptr, timEb, 0, 0);
  gemm(xb,   nullptr, 0, kN, kD,  kD,  in_w, kD, in_b, hA, hb, 0, 0);
  gemm(msgb, nullptr, 0, kE, kED, kED, ee_w, kD, ee_b, nullptr, edgEb, 0, 0);

  float* hcur = hA; float* hnxt = hB;
  for (int l = 0; l < kL; ++l) {
    const LayerP& L = LP[l];
    // node projections
    gemm(hb, nullptr, 0, kN, kD, kD, L.q_w, kD, L.q_b, Qn, nullptr, 0, 0);
    gemm(hb, nullptr, 0, kN, kD, kD, L.k_w, kD, L.k_b, Kn, nullptr, 0, 0);
    gemm(hb, nullptr, 0, kN, kD, kD, L.v_w, kD, L.v_b, Vn, nullptr, 0, 0);
    // edge projections
    gemm(edgEb, nullptr, 0, kE, kD, kD, L.e_w,  kD, L.e_b,  eeP, nullptr, 0, 0);
    gemm(timEb, nullptr, 0, kE, kD, kD, L.tm_w, kD, L.tm_b, teP, nullptr, 0, 0);
    // trust pipeline
    gemm(memhb, dst, 1, kE, kD, kD, L.tq_w, kD, L.tq_b, Qt, nullptr, 0, 0);
    for (int cs = 0; cs < kE; cs += kCH) {
      int mc = kE - cs; if (mc > kCH) mc = kCH;
      gemm(evob, src + cs, 2, mc * kW, kD, kD, L.tk_w, kD, L.tk_b, Ktc, nullptr, 0, 0);
      gemm(evob, src + cs, 2, mc * kW, kD, kD, L.tv_w, kD, L.tv_b, Vtc, nullptr, 0, 0);
      k_trust_attn<<<waves(mc), BS, 0, stream>>>(Qt, Ktc, Vtc, L.log_decay, poolb, cs, mc);
    }
    gemm(poolb, nullptr, 0, kE, kD, kD, L.to1_w, 64, L.to1_b, to1o, nullptr, /*gelu*/1, 0);
    k_trust_fin<<<waves(kE), BS, 0, stream>>>(to1o, L.to2_w, L.to2_b, mem_v, src, dst, trust);
    // segment-softmax attention + aggregation
    k_fill<<<GS, BS, 0, stream>>>(mB, 0.f, (long)kN * kH);  // enc(0) = weakest
    k_fill<<<GS, BS, 0, stream>>>(sB, 0.f, (long)kN * kH);
    k_fill<<<GS, BS, 0, stream>>>(agg, 0.f, (long)kN * kD);
    k_edge_attn<<<waves(kE), BS, 0, stream>>>(Qn, Kn, eeP, teP, trust, src, dst,
                                              attn, (unsigned*)mB);
    k_edge_exp<<<GS, BS, 0, stream>>>(attn, (const unsigned*)mB, sB, dst);
    k_edge_agg<<<waves(kE), BS, 0, stream>>>(attn, sB, Vn, src, dst, agg);
    // fused MLP on concat(agg, mem_h): two-pass accumulate GEMM
    k_cvt<<<GS, BS, 0, stream>>>(agg, aggb, (long)kN * kD);
    gemm(aggb,  nullptr, 0, kN, kD, kD, L.f1_w,           kD, L.f1_b, fu1, nullptr, 0, 0);
    gemm(memhb, nullptr, 0, kN, kD, kD, L.f1_w + kD * kD, kD, nullptr, fu1, fu1b, /*gelu*/1, 1);
    gemm(fu1b, nullptr, 0, kN, kD, kD, L.f2_w, kD, L.f2_b, nullptr, fu2b, 0, 0);
    gemm(fu2b, nullptr, 0, kN, kD, kD, L.o_w,  kD, L.o_b,  fu1, nullptr, 0, 0);
    // gate + residual + layernorm (bf16 dual-write feeds next layer's QKV)
    k_gate_ln<<<waves(kN), BS, 0, stream>>>(hcur, fu1, L.g_w, L.g_b, L.n_g, L.n_b,
                                            hnxt, hb, kN);
    float* tmp = hcur; hcur = hnxt; hnxt = tmp;
  }
  // final layernorm -> d_out
  k_gate_ln<<<waves(kN), BS, 0, stream>>>(hcur, nullptr, nullptr, nullptr,
                                          fn_g, fn_b, (float*)d_out, kN ? nullptr : nullptr, kN);
}
